// GATNet_Simple_34626026340855
// MI455X (gfx1250) — matrix operand order repaired and checked by reading, not compiled, not run on hardware
//
#include <hip/hip_runtime.h>
#include <hip/hip_bf16.h>
#include <cstdint>
#include <cstddef>

// ---------------------------------------------------------------------------
// Types for CDNA5 WMMA (wave32): v_wmma_f32_16x16x32_bf16
// ---------------------------------------------------------------------------
typedef __bf16 bf16_t;
typedef __attribute__((ext_vector_type(16))) bf16_t v16bf;
typedef __attribute__((ext_vector_type(8)))  float  v8f;

#define DEV __device__ __forceinline__

DEV unsigned short f32_to_bf16u(float f) {
    union { float f; unsigned u; } v; v.f = f;
    unsigned lsb = (v.u >> 16) & 1u;
    unsigned r = v.u + 0x7FFFu + lsb;           // round-to-nearest-even
    return (unsigned short)(r >> 16);
}
DEV float bf16u_to_f32(unsigned short h) {
    union { unsigned u; float f; } v; v.u = ((unsigned)h) << 16; return v.f;
}
// monotone uint encoding of float (for atomicMax over arbitrary-sign floats)
DEV unsigned f32_ord(float f) {
    union { float f; unsigned u; } v; v.f = f;
    return (v.u & 0x80000000u) ? ~v.u : (v.u | 0x80000000u);
}
DEV float ord_f32(unsigned k) {
    union { unsigned u; float f; } v;
    v.u = (k & 0x80000000u) ? (k & 0x7FFFFFFFu) : ~k;
    return v.f;
}

// ---------------------------------------------------------------------------
// Utility fill
// ---------------------------------------------------------------------------
__global__ void k_fill_u32(unsigned* __restrict__ p, unsigned v, long n) {
    long i = (long)blockIdx.x * blockDim.x + threadIdx.x;
    if (i < n) p[i] = v;
}

// ---------------------------------------------------------------------------
// gat1 GEMM: [N,4] @ [4,256] -> bf16 [N,256]  (K=4: scalar FMAs)
// ---------------------------------------------------------------------------
__global__ void k_gemm_small(const float* __restrict__ x, const float* __restrict__ W,
                             unsigned short* __restrict__ hb, int Nn, int K, int F) {
    long i = (long)blockIdx.x * blockDim.x + threadIdx.x;
    if (i >= (long)Nn * F) return;
    int n = (int)(i / F), c = (int)(i % F);
    float s = 0.f;
    for (int k = 0; k < K; ++k) s += x[(long)n * K + k] * W[(long)k * F + c];
    hb[i] = f32_to_bf16u(s);
}

// ---------------------------------------------------------------------------
// Pack f32 weights [K,F] into WMMA-B-fragment-major bf16:
//   Wp[((tn*(K/32)+kc)*32 + lane)*16 + e] = bf16( W[(kc*32+(lane>>4)*16+e)*F + tn*16+(lane&15)] )
// so each lane's 16 fragment elements are contiguous (two uint4 loads).
// ---------------------------------------------------------------------------
__global__ void k_pack_w(const float* __restrict__ W, unsigned short* __restrict__ Wp,
                         int K, int F) {
    int i = blockIdx.x * blockDim.x + threadIdx.x;
    if (i >= K * F) return;
    int e     = i & 15;
    int lane  = (i >> 4) & 31;
    int chunk = i >> 9;                // tn*(K/32) + kc
    int KC = K >> 5;
    int tn = chunk / KC, kc = chunk % KC;
    int k   = kc * 32 + (lane >> 4) * 16 + e;
    int col = tn * 16 + (lane & 15);
    Wp[i] = f32_to_bf16u(W[(long)k * F + col]);
}

// ---------------------------------------------------------------------------
// WMMA bf16 GEMM, K fixed at compile time (=256): D[M,F] = A[M,256] @ W[256,F]
// A bf16 row-major, Wp pre-packed fragment-major bf16, D bf16.
// One wave per 16x16 tile; fully unrolled -> 8x v_wmma_f32_16x16x32_bf16.
// blockDim = 128 (4 waves = 4 consecutive M tiles), grid = (ceil(M/64), F/16)
// ---------------------------------------------------------------------------
template <int K>
__global__ __launch_bounds__(128)
void k_gemm_wmma_bf16(const unsigned short* __restrict__ A,
                      const unsigned short* __restrict__ Wp,
                      unsigned short* __restrict__ D,
                      int M, int F) {
    constexpr int KC = K / 32;
    int lane = threadIdx.x & 31;
    int wave = threadIdx.x >> 5;
    int tm = blockIdx.x * 4 + wave;     // M tile (wave-uniform)
    int tn = blockIdx.y;                // N tile
    if (tm * 16 >= M) return;           // uniform per wave -> EXEC stays all-1s
    int lh  = lane >> 4;                // 0 or 1 (half-wave)
    int l15 = lane & 15;
    int row = tm * 16 + l15;
    int rc  = row < M ? row : M - 1;    // clamp (keeps EXEC full for WMMA)
    int col = tn * 16 + l15;

    const unsigned short* arow = A + (long)rc * K;
    const uint4* wq = reinterpret_cast<const uint4*>(Wp + (long)tn * KC * 512 + lane * 16);

    v8f acc = {};
    #pragma unroll
    for (int kc = 0; kc < KC; ++kc) {
        // A fragment: lane<16 holds K {0..15}, lane>=16 holds {16..31}
        union { v16bf v; uint4 q[2]; } af;
        const uint4* pa = reinterpret_cast<const uint4*>(arow + kc * 32 + lh * 16);
        af.q[0] = pa[0];                // 8 contiguous bf16
        af.q[1] = pa[1];                // +8 elements
        // B fragment: pre-packed, 2 contiguous uint4 per lane per k-chunk
        union { v16bf v; uint4 q[2]; } bfr;
        bfr.q[0] = wq[kc * 64 + 0];     // 512 ushorts per chunk = 64 uint4
        bfr.q[1] = wq[kc * 64 + 1];
        acc = __builtin_amdgcn_wmma_f32_16x16x32_bf16(
            false, af.v, false, bfr.v, (short)0, acc, false, false);
    }
    // C/D layout: element r -> row tm*16 + r + lh*8, col = tn*16 + l15
    if (tm * 16 + 16 <= M) {            // wave-uniform fast path: no guards
        #pragma unroll
        for (int r = 0; r < 8; ++r)
            D[(long)(tm * 16 + r + lh * 8) * F + col] = f32_to_bf16u(acc[r]);
    } else {
        #pragma unroll
        for (int r = 0; r < 8; ++r) {
            int rm = tm * 16 + r + lh * 8;
            if (rm < M) D[(long)rm * F + col] = f32_to_bf16u(acc[r]);
        }
    }
}

// ---------------------------------------------------------------------------
// attention scores: asrc[n,h] = sum_c h[n,h,c]*a_src[h,c]; same for adst
// ---------------------------------------------------------------------------
__global__ void k_attn_scores(const unsigned short* __restrict__ hb,
                              const float* __restrict__ a_src, const float* __restrict__ a_dst,
                              float* __restrict__ asrc, float* __restrict__ adst,
                              int Nn, int H, int C) {
    int i = blockIdx.x * blockDim.x + threadIdx.x;
    if (i >= Nn * H) return;
    int n = i / H, h = i % H;
    int F = H * C;
    const unsigned short* p = hb + (long)n * F + h * C;
    float s = 0.f, d = 0.f;
    for (int c = 0; c < C; ++c) {
        float v = bf16u_to_f32(p[c]);
        s += v * a_src[h * C + c];
        d += v * a_dst[h * C + c];
    }
    asrc[i] = s; adst[i] = d;
}

// ---------------------------------------------------------------------------
// edge pass 1: logits + segment max (self-loops appended after E real edges)
// ---------------------------------------------------------------------------
__global__ void k_edge_logit_max(const int* __restrict__ ei, int E, int Nn, int H,
                                 const float* __restrict__ asrc, const float* __restrict__ adst,
                                 float* __restrict__ alpha, unsigned* __restrict__ segmax) {
    long i = (long)blockIdx.x * blockDim.x + threadIdx.x;
    long tot = (long)(E + Nn) * H;
    if (i >= tot) return;
    long e = i / H; int h = (int)(i % H);
    int s, d;
    if (e < E) { s = ei[e]; d = ei[(long)E + e]; } else { s = d = (int)(e - E); }
    float l = asrc[(long)s * H + h] + adst[(long)d * H + h];
    l = l > 0.f ? l : 0.2f * l;          // leaky_relu(0.2)
    alpha[i] = l;
    atomicMax(&segmax[(long)d * H + h], f32_ord(l));
}

// ---------------------------------------------------------------------------
// edge pass 2: ex = exp(logit - max[dst]); segment sum
// ---------------------------------------------------------------------------
__global__ void k_edge_expsum(const int* __restrict__ ei, int E, int Nn, int H,
                              const unsigned* __restrict__ segmax,
                              float* __restrict__ alpha, float* __restrict__ segsum) {
    long i = (long)blockIdx.x * blockDim.x + threadIdx.x;
    long tot = (long)(E + Nn) * H;
    if (i >= tot) return;
    long e = i / H; int h = (int)(i % H);
    int d;
    if (e < E) { d = ei[(long)E + e]; } else { d = (int)(e - E); }
    float m = ord_f32(segmax[(long)d * H + h]);
    float ex = __expf(alpha[i] - m);
    alpha[i] = ex;
    atomicAdd(&segsum[(long)d * H + h], ex);
}

// ---------------------------------------------------------------------------
// edge pass 3: out[dst] += h[src] * alpha   (4 features per thread, bf16 gather)
// ---------------------------------------------------------------------------
__global__ void k_edge_aggr(const int* __restrict__ ei, int E, int Nn, int H, int C,
                            const unsigned short* __restrict__ hb,
                            const float* __restrict__ alpha,
                            const float* __restrict__ segsum, float* __restrict__ agg) {
    int F = H * C;
    int q = F >> 2;
    long i = (long)blockIdx.x * blockDim.x + threadIdx.x;
    long tot = (long)(E + Nn) * q;
    if (i >= tot) return;
    long e = i / q; int f = (int)(i % q) * 4;
    int h = f / C;
    int s, d;
    if (e < E) { s = ei[e]; d = ei[(long)E + e]; } else { s = d = (int)(e - E); }
    float a = alpha[e * H + h] / (segsum[(long)d * H + h] + 1e-16f);
    uint2 raw = *reinterpret_cast<const uint2*>(hb + (long)s * F + f);
    float* o = agg + (long)d * F + f;
    atomicAdd(&o[0], bf16u_to_f32((unsigned short)(raw.x & 0xFFFFu)) * a);
    atomicAdd(&o[1], bf16u_to_f32((unsigned short)(raw.x >> 16)) * a);
    atomicAdd(&o[2], bf16u_to_f32((unsigned short)(raw.y & 0xFFFFu)) * a);
    atomicAdd(&o[3], bf16u_to_f32((unsigned short)(raw.y >> 16)) * a);
}

// ---------------------------------------------------------------------------
// bias + activation (mode 0 = elu, 1 = relu), write bf16 next-layer input
// ---------------------------------------------------------------------------
__global__ void k_bias_act(const float* __restrict__ agg, const float* __restrict__ b,
                           unsigned short* __restrict__ xb, long n, int F, int mode) {
    long i = (long)blockIdx.x * blockDim.x + threadIdx.x;
    if (i >= n) return;
    float v = agg[i] + b[i % F];
    v = mode ? (v > 0.f ? v : 0.f) : (v > 0.f ? v : __expf(v) - 1.f);
    xb[i] = f32_to_bf16u(v);
}

// ---------------------------------------------------------------------------
// global max pool over graphs (values >= 0 post-relu -> plain uint atomicMax)
// ---------------------------------------------------------------------------
__global__ void k_pool_max(const unsigned short* __restrict__ xb,
                           const int* __restrict__ batch, float* __restrict__ pooled, int Nn) {
    long i = (long)blockIdx.x * blockDim.x + threadIdx.x;
    if (i >= (long)Nn * 64) return;
    int n = (int)(i >> 6), c = (int)(i & 63);
    float v = bf16u_to_f32(xb[i]);
    atomicMax((unsigned*)&pooled[(long)batch[n] * 64 + c], __float_as_uint(v));
}

// ---------------------------------------------------------------------------
// xg = pooled @ fcg_w + fcg_b   ([G,64]x[64,64], tiny)
// ---------------------------------------------------------------------------
__global__ void k_fc64(const float* __restrict__ in, const float* __restrict__ w,
                       const float* __restrict__ b, float* __restrict__ out, int G) {
    int i = blockIdx.x * blockDim.x + threadIdx.x;
    if (i >= G * 64) return;
    int g = i >> 6, c = i & 63;
    float s = b[c];
    for (int k = 0; k < 64; ++k) s += in[(long)g * 64 + k] * w[k * 64 + c];
    out[i] = s;
}

// ---------------------------------------------------------------------------
// CNN branch fused: conv1d(5->64,k3,pad1) + relu + maxpool(L) + FC(64->64)
// One workgroup per graph; target slice staged in LDS (20KB).
// ---------------------------------------------------------------------------
__global__ void k_cnn(const float* __restrict__ target, const float* __restrict__ cw,
                      const float* __restrict__ cb, const float* __restrict__ fw,
                      const float* __restrict__ fb, float* __restrict__ xt, int L) {
    __shared__ float tg[5000];
    __shared__ float xm[64];
    int g = blockIdx.x, t = threadIdx.x;
    const float* tp = target + (long)g * L * 5;
    for (int i = t; i < L * 5; i += 64) tg[i] = tp[i];
    __syncthreads();
    float w[15];
    #pragma unroll
    for (int j = 0; j < 15; ++j) w[j] = cw[t * 15 + j];
    float bb = cb[t];
    float m = 0.f;                        // relu output >= 0
    for (int l = 0; l < L; ++l) {
        float s = bb;
        #pragma unroll
        for (int k = 0; k < 3; ++k) {
            int ll = l + k - 1;
            if (ll >= 0 && ll < L) {
                #pragma unroll
                for (int ic = 0; ic < 5; ++ic) s += tg[ll * 5 + ic] * w[ic * 3 + k];
            }
        }
        s = s > 0.f ? s : 0.f;
        m = s > m ? s : m;
    }
    xm[t] = m;
    __syncthreads();
    float s = fb[t];
    for (int k = 0; k < 64; ++k) s += xm[k] * fw[k * 64 + t];
    xt[(long)g * 64 + t] = s;
}

// ---------------------------------------------------------------------------
// head: relu(concat(xg,xt) @ fus_w + fus_b) @ out_w + out_b
// ---------------------------------------------------------------------------
__global__ void k_head(const float* __restrict__ xg, const float* __restrict__ xt,
                       const float* __restrict__ fw, const float* __restrict__ fb,
                       const float* __restrict__ ow, const float* __restrict__ ob,
                       float* __restrict__ out) {
    __shared__ float xc[128];
    __shared__ float y[64];
    int g = blockIdx.x, t = threadIdx.x;
    xc[t]      = xg[(long)g * 64 + t];
    xc[64 + t] = xt[(long)g * 64 + t];
    __syncthreads();
    float s = fb[t];
    for (int i = 0; i < 128; ++i) s += xc[i] * fw[i * 64 + t];
    y[t] = s > 0.f ? s : 0.f;
    __syncthreads();
    if (t < 2) {
        float o = ob[t];
        for (int c = 0; c < 64; ++c) o += y[c] * ow[c * 2 + t];
        out[(long)g * 2 + t] = o;
    }
}

// ---------------------------------------------------------------------------
// launcher
// ---------------------------------------------------------------------------
static inline unsigned cdiv(long a, long b) { return (unsigned)((a + b - 1) / b); }

extern "C" void kernel_launch(void* const* d_in, const int* in_sizes, int n_in,
                              void* d_out, int out_size, void* d_ws, size_t ws_size,
                              hipStream_t stream) {
    const float* x      = (const float*)d_in[0];
    const int*   ei     = (const int*)d_in[1];
    const int*   batch  = (const int*)d_in[2];
    const float* target = (const float*)d_in[3];
    const float* W1 = (const float*)d_in[4];
    const float* as1 = (const float*)d_in[5];
    const float* ad1 = (const float*)d_in[6];
    const float* b1 = (const float*)d_in[7];
    const float* W2 = (const float*)d_in[8];
    const float* as2 = (const float*)d_in[9];
    const float* ad2 = (const float*)d_in[10];
    const float* b2 = (const float*)d_in[11];
    const float* W3 = (const float*)d_in[12];
    const float* as3 = (const float*)d_in[13];
    const float* ad3 = (const float*)d_in[14];
    const float* b3 = (const float*)d_in[15];
    const float* fcg_w = (const float*)d_in[16];
    const float* fcg_b = (const float*)d_in[17];
    const float* conv_w = (const float*)d_in[18];
    const float* conv_b = (const float*)d_in[19];
    const float* fcxt_w = (const float*)d_in[20];
    const float* fcxt_b = (const float*)d_in[21];
    const float* fus_w = (const float*)d_in[22];
    const float* fus_b = (const float*)d_in[23];
    const float* out_w = (const float*)d_in[24];
    const float* out_b = (const float*)d_in[25];
    float* out = (float*)d_out;

    const int Nn = in_sizes[2];
    const int E  = in_sizes[1] / 2;
    const int G  = out_size / 2;
    const int L  = in_sizes[3] / (5 * G);
    const long Et = (long)E + Nn;

    // workspace carve-out
    char* ws = (char*)d_ws;
    auto carve = [&](size_t bytes) { char* p = ws; ws += (bytes + 255) & ~(size_t)255; return p; };
    unsigned short* hbf = (unsigned short*)carve((size_t)Nn * 256 * 2); // GEMM output (bf16)
    unsigned short* xbf = (unsigned short*)carve((size_t)Nn * 256 * 2); // activated input (bf16)
    float*    agg    = (float*)carve((size_t)Nn * 256 * 4);             // aggregation (fp32, L2-resident)
    float*    asrc   = (float*)carve((size_t)Nn * 4 * 4);
    float*    adst   = (float*)carve((size_t)Nn * 4 * 4);
    unsigned* segmax = (unsigned*)carve((size_t)Nn * 4 * 4);
    float*    segsum = (float*)carve((size_t)Nn * 4 * 4);
    float*    alpha  = (float*)carve((size_t)Et * 4 * 4);
    float*    pooled = (float*)carve((size_t)G * 64 * 4);
    float*    xg     = (float*)carve((size_t)G * 64 * 4);
    float*    xt     = (float*)carve((size_t)G * 64 * 4);
    unsigned short* Wp2 = (unsigned short*)carve((size_t)256 * 256 * 2); // packed W2 (bf16)
    unsigned short* Wp3 = (unsigned short*)carve((size_t)256 * 64 * 2);  // packed W3 (bf16)

    const int BT = 256;

    // pack weights into WMMA-B-fragment layout (tiny, once per launch)
    k_pack_w<<<cdiv(256 * 256, BT), BT, 0, stream>>>(W2, Wp2, 256, 256);
    k_pack_w<<<cdiv(256 * 64, BT), BT, 0, stream>>>(W3, Wp3, 256, 64);

    auto attention = [&](int H, int C, const float* a_s, const float* a_d,
                         const float* bias, int mode) {
        int F = H * C;
        // clear segment buffers + accumulator
        k_fill_u32<<<cdiv((long)Nn * H, BT), BT, 0, stream>>>(segmax, 0u, (long)Nn * H);
        k_fill_u32<<<cdiv((long)Nn * H, BT), BT, 0, stream>>>((unsigned*)segsum, 0u, (long)Nn * H);
        k_fill_u32<<<cdiv((long)Nn * F, BT), BT, 0, stream>>>((unsigned*)agg, 0u, (long)Nn * F);
        k_attn_scores<<<cdiv((long)Nn * H, BT), BT, 0, stream>>>(hbf, a_s, a_d, asrc, adst, Nn, H, C);
        k_edge_logit_max<<<cdiv(Et * H, BT), BT, 0, stream>>>(ei, E, Nn, H, asrc, adst, alpha, segmax);
        k_edge_expsum<<<cdiv(Et * H, BT), BT, 0, stream>>>(ei, E, Nn, H, segmax, alpha, segsum);
        k_edge_aggr<<<cdiv(Et * (F / 4), BT), BT, 0, stream>>>(ei, E, Nn, H, C, hbf, alpha, segsum, agg);
        k_bias_act<<<cdiv((long)Nn * F, BT), BT, 0, stream>>>(agg, bias, xbf, (long)Nn * F, F, mode);
    };

    // ---- layer 1: [N,4] @ W1 -> [N,256], H=4, C=64, elu
    k_gemm_small<<<cdiv((long)Nn * 256, BT), BT, 0, stream>>>(x, W1, hbf, Nn, 4, 256);
    attention(4, 64, as1, ad1, b1, /*elu*/0);

    // ---- layer 2: [N,256] @ W2 -> [N,256] via WMMA (8 unrolled wmma), H=2, C=128, elu
    {
        dim3 grid(cdiv(Nn, 64), 256 / 16);
        k_gemm_wmma_bf16<256><<<grid, 128, 0, stream>>>(xbf, Wp2, hbf, Nn, 256);
    }
    attention(2, 128, as2, ad2, b2, /*elu*/0);

    // ---- layer 3: [N,256] @ W3 -> [N,64] via WMMA, H=1, C=64, relu
    {
        dim3 grid(cdiv(Nn, 64), 64 / 16);
        k_gemm_wmma_bf16<256><<<grid, 128, 0, stream>>>(xbf, Wp3, hbf, Nn, 64);
    }
    attention(1, 64, as3, ad3, b3, /*relu*/1);

    // ---- global max pool + fcg
    k_fill_u32<<<cdiv((long)G * 64, BT), BT, 0, stream>>>((unsigned*)pooled, 0u, (long)G * 64);
    k_pool_max<<<cdiv((long)Nn * 64, BT), BT, 0, stream>>>(xbf, batch, pooled, Nn);
    k_fc64<<<cdiv((long)G * 64, BT), BT, 0, stream>>>(pooled, fcg_w, fcg_b, xg, G);

    // ---- CNN branch (fused conv+relu+maxpool+fc through LDS)
    k_cnn<<<G, 64, 0, stream>>>(target, conv_w, conv_b, fcxt_w, fcxt_b, xt, L);

    // ---- fusion head
    k_head<<<G, 64, 0, stream>>>(xg, xt, fus_w, fus_b, out_w, out_b, out);
}
